// LowProcessBlock_13151189860679
// MI455X (gfx1250) — matrix operand order repaired and checked by reading, not compiled, not run on hardware
//
#include <hip/hip_runtime.h>
#include <hip/hip_bf16.h>

// SS2D block for MI455X (gfx1250, wave32).
//  - GEMMs on v_wmma_f32_16x16x32_f16 (f16 in, f32 accum).
//  - W_in / W_out staged into LDS via the Tensor Data Mover (TDM) with row
//    padding (bank-conflict-free B-fragment reads), s_wait_tensorcnt sync.
//  - Bs/Cs stored [B,K,L,DS] so the sequential scan reads contiguous lines.
//  - Scan: one wave per (b,k,d) channel, lanes = DS states, shfl_xor reduce.

typedef __attribute__((ext_vector_type(16))) _Float16 v16h;
typedef __attribute__((ext_vector_type(8)))  float    v8f;
typedef __attribute__((ext_vector_type(4)))  unsigned int u32x4;
typedef __attribute__((ext_vector_type(8)))  int      i32x8;
typedef __attribute__((ext_vector_type(4)))  int      i32x4;

#define BB  2
#define HH  256
#define WW  129
#define DMM 64
#define DII 128
#define DSN 16
#define DTRK 4
#define KD  2
#define LL  (HH * WW)            // 33024
#define NROWS (BB * LL)          // 66048  (divisible by 16)
#define NPROJ (DTRK + 2 * DSN)   // 36

// ---------------------------------------------------------------------------
// TDM helper: DMA n_elems f32 (n_elems < 65536) from global into LDS as a
// 1-row tile, inserting 1 DWORD of padding every 2^(pad_code+1) DWORDs so the
// LDS row stride is odd (conflict-free column reads).  D# layout per ISA §8.
// Issued by one wave; includes the tensorcnt wait.
// ---------------------------------------------------------------------------
__device__ __forceinline__ void tdm_load_f32(const float* gptr, unsigned lds_off,
                                             unsigned n_elems, int pad_code) {
  const unsigned long long ga = (unsigned long long)gptr;
  u32x4 g0;
  g0[0] = 1u;                                               // count=1 (valid)
  g0[1] = lds_off;                                          // lds_addr (bytes)
  g0[2] = (unsigned)ga;                                     // global_addr[31:0]
  g0[3] = (unsigned)((ga >> 32) & 0x01ffffffu) | (2u << 30);// addr[56:32], type=2
  i32x8 g1;
  g1[0] = (2 << 16) | (1 << 20) | (pad_code << 22);         // 4B, pad_en, interval
  g1[1] = (int)(n_elems << 16);                             // tensor_dim0[15:0]
  g1[2] = (1 << 16);                                        // tensor_dim1 = 1
  g1[3] = (int)(n_elems << 16);                             // tile_dim0
  g1[4] = 1;                                                // tile_dim1 = 1
  g1[5] = (int)n_elems;                                     // tensor_dim0_stride
  g1[6] = 0; g1[7] = 0;
  i32x4 z4 = {0, 0, 0, 0};
#if __clang_major__ >= 23
  i32x8 z8 = {0, 0, 0, 0, 0, 0, 0, 0};
  __builtin_amdgcn_tensor_load_to_lds(g0, g1, z4, z4, z8, 0);
#else
  __builtin_amdgcn_tensor_load_to_lds(g0, g1, z4, z4, 0);
#endif
  __builtin_amdgcn_s_wait_tensorcnt(0);
}

// ---------------------------------------------------------------------------
// 1) in_proj: xz[r, 0:256] = x[r, 0:64] @ W_in^T.  One wave per 16-row tile,
//    K=64 -> 2 WMMA k-chunks, N=256 -> 16 n-tiles (32 WMMAs / wave).
//    W_in staged in LDS by TDM, row stride 65 DWORDs (bank-conflict-free).
//    o<128 -> xc in NCHW [B,DI,L] (contiguous float4 stores); o>=128 -> z.
// ---------------------------------------------------------------------------
__global__ void k_inproj(const float* __restrict__ x, const float* __restrict__ Win,
                         float* __restrict__ xc, float* __restrict__ zb) {
  __shared__ float sW[256 * 65];            // 256 rows x (64 + 1 pad) DWORDs
  if (threadIdx.x < 32)
    tdm_load_f32(Win, (unsigned)(unsigned long long)&sW[0], 256 * 64, /*every 64dw*/5);
  __syncthreads();

  const int tid  = blockIdx.x * blockDim.x + threadIdx.x;
  const int wave = tid >> 5, lane = tid & 31;
  if (wave >= NROWS / 16) return;
  const int mlo = lane & 15, hi = lane >> 4;
  const int row0 = wave * 16, row = row0 + mlo;
  const int b = row0 / LL;                  // 16-row tile never straddles b
  const int pbase = row0 - b * LL;
  const float* xr = x + (size_t)row * DMM;

  v16h a[2];
  #pragma unroll
  for (int kc = 0; kc < 2; ++kc) {
    const int c0 = kc * 32 + hi * 16;
    #pragma unroll
    for (int e = 0; e < 8; ++e) {
      a[kc][e]     = (_Float16)xr[c0 + e];
      a[kc][e + 8] = (_Float16)xr[c0 + 8 + e];
    }
  }
  // n-tiles 0..7: xc half (o < 128), contiguous along p -> float4 stores
  for (int nt = 0; nt < 8; ++nt) {
    const int o = nt * 16 + mlo;
    v8f acc = {};
    #pragma unroll
    for (int kc = 0; kc < 2; ++kc) {
      const float* wr = &sW[o * 65 + kc * 32 + hi * 16];
      v16h bf;
      #pragma unroll
      for (int e = 0; e < 16; ++e) bf[e] = (_Float16)wr[e];
      acc = __builtin_amdgcn_wmma_f32_16x16x32_f16(false, a[kc], false, bf,
                                                   (short)0, acc, false, false);
    }
    float* xcp = xc + ((size_t)b * DII + o) * LL + pbase + hi * 8;
    float4 v0 = make_float4(acc[0], acc[1], acc[2], acc[3]);
    float4 v1 = make_float4(acc[4], acc[5], acc[6], acc[7]);
    *(float4*)(xcp)     = v0;
    *(float4*)(xcp + 4) = v1;
  }
  // n-tiles 8..15: z half (o >= 128), rows strided by DII
  for (int nt = 8; nt < 16; ++nt) {
    const int o = nt * 16 + mlo;
    v8f acc = {};
    #pragma unroll
    for (int kc = 0; kc < 2; ++kc) {
      const float* wr = &sW[o * 65 + kc * 32 + hi * 16];
      v16h bf;
      #pragma unroll
      for (int e = 0; e < 16; ++e) bf[e] = (_Float16)wr[e];
      acc = __builtin_amdgcn_wmma_f32_16x16x32_f16(false, a[kc], false, bf,
                                                   (short)0, acc, false, false);
    }
    float* zp = zb + (size_t)(row0 + hi * 8) * DII + (o - DII);
    #pragma unroll
    for (int r = 0; r < 8; ++r) zp[(size_t)r * DII] = acc[r];
  }
}

// ---------------------------------------------------------------------------
// 2) Depthwise 3x3 conv, pad 1, + bias.  xc [B,DI,H,W] -> xf [B,DI,L].
// ---------------------------------------------------------------------------
__global__ void k_conv(const float* __restrict__ xc, const float* __restrict__ cw,
                       const float* __restrict__ cb, float* __restrict__ xf) {
  const long long i = (long long)blockIdx.x * blockDim.x + threadIdx.x;
  if (i >= (long long)BB * DII * LL) return;
  const int p  = (int)(i % LL);
  const int bd = (int)(i / LL);
  const int d  = bd % DII;
  const int h = p / WW, w = p % WW;
  const float* src = xc + (size_t)bd * LL;
  const float* k9  = cw + d * 9;
  float s = cb[d];
  #pragma unroll
  for (int dh = -1; dh <= 1; ++dh) {
    const int hh = h + dh;
    if (hh < 0 || hh >= HH) continue;
    #pragma unroll
    for (int dw = -1; dw <= 1; ++dw) {
      const int ww2 = w + dw;
      if (ww2 < 0 || ww2 >= WW) continue;
      s += k9[(dh + 1) * 3 + (dw + 1)] * src[hh * WW + ww2];
    }
  }
  xf[i] = s;
}

// ---------------------------------------------------------------------------
// 3) x_dbl = x_proj_weight[k] (36x128, padded to 48) @ xs (128xL).  The
//    Fourier gather + direction flip is fused into the B-operand addressing.
//    Rows 0..3 -> dts_raw [B,K,4,L]; 4..19 -> Bs [B,K,L,DS]; 20..35 -> Cs.
// ---------------------------------------------------------------------------
__global__ void k_xdbl(const float* __restrict__ xf, const float* __restrict__ xpw,
                       const int* __restrict__ freq, float* __restrict__ dtsr,
                       float* __restrict__ Bsw, float* __restrict__ Csw) {
  const int tid  = blockIdx.x * blockDim.x + threadIdx.x;
  const int wave = tid >> 5, lane = tid & 31;
  const int NT = LL / 16;                 // 2064
  if (wave >= BB * KD * 3 * NT) return;
  int w = wave;
  const int nt = w % NT;  w /= NT;
  const int mt = w % 3;   w /= 3;
  const int kd = w % KD;  const int b = w / KD;
  const int mlo = lane & 15, hi = lane >> 4;

  const int m = mt * 16 + mlo;
  const bool mv = (m < NPROJ);
  const float* wrow = xpw + ((size_t)kd * NPROJ + (mv ? m : 0)) * DII;
  v16h a[4];
  #pragma unroll
  for (int kc = 0; kc < 4; ++kc) {
    const int c0 = kc * 32 + hi * 16;
    #pragma unroll
    for (int e = 0; e < 8; ++e) {
      a[kc][e]     = mv ? (_Float16)wrow[c0 + e]      : (_Float16)0.f;
      a[kc][e + 8] = mv ? (_Float16)wrow[c0 + 8 + e] : (_Float16)0.f;
    }
  }
  const int l = nt * 16 + mlo;            // scan-step column
  const int s = kd ? (LL - 1 - l) : l;    // direction flip
  const int g = freq[s];                  // Fourier gather
  const float* colb = xf + (size_t)b * DII * LL + g;
  v8f acc = {};
  #pragma unroll
  for (int kc = 0; kc < 4; ++kc) {
    const int d0 = kc * 32 + hi * 16;
    v16h bf;
    #pragma unroll
    for (int e = 0; e < 16; ++e) bf[e] = (_Float16)colb[(size_t)(d0 + e) * LL];
    acc = __builtin_amdgcn_wmma_f32_16x16x32_f16(false, a[kc], false, bf,
                                                 (short)0, acc, false, false);
  }
  const size_t bk = (size_t)b * KD + kd;
  #pragma unroll
  for (int r = 0; r < 8; ++r) {
    const int mo = mt * 16 + hi * 8 + r;
    if (mo >= NPROJ) continue;
    const float v = acc[r];
    if (mo < DTRK)            dtsr[(bk * DTRK + mo) * LL + l] = v;
    else if (mo < DTRK + DSN) Bsw[(bk * LL + l) * DSN + (mo - DTRK)] = v;
    else                      Csw[(bk * LL + l) * DSN + (mo - DTRK - DSN)] = v;
  }
}

// ---------------------------------------------------------------------------
// 4) delta = softplus(dt_projs_weight[k] @ dts + bias)   [B,K,DI,L]
// ---------------------------------------------------------------------------
__global__ void k_delta(const float* __restrict__ dtsr, const float* __restrict__ dtw,
                        const float* __restrict__ dtb, float* __restrict__ dlt) {
  const long long i = (long long)blockIdx.x * blockDim.x + threadIdx.x;
  if (i >= (long long)BB * KD * DII * LL) return;
  const int l = (int)(i % LL);
  const int d = (int)((i / LL) % DII);
  const int k = (int)((i / ((long long)LL * DII)) % KD);
  const int b = (int)(i / ((long long)LL * DII * KD));
  const size_t bk = (size_t)b * KD + k;
  float acc = dtb[k * DII + d];
  #pragma unroll
  for (int r = 0; r < DTRK; ++r)
    acc += dtw[((size_t)k * DII + d) * DTRK + r] * dtsr[(bk * DTRK + r) * LL + l];
  dlt[i] = (acc > 20.f) ? acc : log1pf(__expf(acc));
}

// 5) inverse permutation of the Fourier sort
__global__ void k_inv(const int* __restrict__ freq, int* __restrict__ inv) {
  const int i = blockIdx.x * blockDim.x + threadIdx.x;
  if (i < LL) inv[freq[i]] = i;
}

// ---------------------------------------------------------------------------
// 6) Selective scan.  One wave per (b,k,d) channel; lanes 0..15 hold the DS
//    states (lanes 16..31 mirror).  h = exp(dl*A)*h + (dl*u)*B ; y = <h,C>.
//    shfl_xor reduction (DS-permute path); prefetch ahead on the serial stream.
// ---------------------------------------------------------------------------
__global__ void k_scan(const float* __restrict__ xf, const float* __restrict__ dlt,
                       const float* __restrict__ Bsw, const float* __restrict__ Csw,
                       const float* __restrict__ Alogs, const float* __restrict__ Dsv,
                       const int* __restrict__ freq, float* __restrict__ ydir) {
  const int tid  = blockIdx.x * blockDim.x + threadIdx.x;
  const int wave = tid >> 5, lane = tid & 31;
  if (wave >= BB * KD * DII) return;
  const int n = lane & 15;
  const int d = wave % DII;
  const int k = (wave / DII) % KD;
  const int b = wave / (DII * KD);
  const int ch = k * DII + d;
  const float An = -__expf(Alogs[(size_t)ch * DSN + n]);
  const float Dv = Dsv[ch];
  const size_t bk = (size_t)b * KD + k;
  const float* drow = dlt + (bk * DII + d) * LL;
  const float* Brow = Bsw + bk * LL * DSN;
  const float* Crow = Csw + bk * LL * DSN;
  const float* xrow = xf + ((size_t)b * DII + d) * LL;
  float* yrow = ydir + (bk * DII + d) * LL;

  float h = 0.f;
  for (int t = 0; t < LL; ++t) {
    if (t + 16 < LL) {
      __builtin_prefetch(drow + t + 16, 0, 1);
      __builtin_prefetch(Brow + (size_t)(t + 16) * DSN, 0, 1);
      __builtin_prefetch(Crow + (size_t)(t + 16) * DSN, 0, 1);
    }
    const int   s  = k ? (LL - 1 - t) : t;
    const float u  = xrow[freq[s]];
    const float dl = drow[t];
    const float Bn = Brow[(size_t)t * DSN + n];
    const float Cn = Crow[(size_t)t * DSN + n];
    h = __expf(dl * An) * h + (dl * u) * Bn;
    float y = h * Cn;
    y += __shfl_xor(y, 1, 32);
    y += __shfl_xor(y, 2, 32);
    y += __shfl_xor(y, 4, 32);
    y += __shfl_xor(y, 8, 32);
    if (lane == 0) yrow[s] = y + Dv * u;   // write at sorted position
  }
}

// ---------------------------------------------------------------------------
// 7) Combine dirs + inverse-permute + LayerNorm(DI) + SiLU gate -> f16 rows.
//    One wave per (b,p) row; 4 channels per lane.
// ---------------------------------------------------------------------------
__global__ void k_lngate(const float* __restrict__ ydir, const float* __restrict__ zb,
                         const int* __restrict__ inv, const float* __restrict__ lnw,
                         const float* __restrict__ lnb, _Float16* __restrict__ yg) {
  const int tid  = blockIdx.x * blockDim.x + threadIdx.x;
  const int wave = tid >> 5, lane = tid & 31;
  if (wave >= NROWS) return;
  const int b = wave / LL, p = wave % LL;
  const int ls = inv[p];
  const float* y0 = ydir + ((size_t)b * KD + 0) * DII * LL + ls;
  const float* y1 = ydir + ((size_t)b * KD + 1) * DII * LL + ls;
  float v[4], sum = 0.f, sq = 0.f;
  #pragma unroll
  for (int j = 0; j < 4; ++j) {
    const int d = lane + 32 * j;
    const float t = y0[(size_t)d * LL] + y1[(size_t)d * LL];
    v[j] = t; sum += t; sq += t * t;
  }
  #pragma unroll
  for (int m = 1; m < 32; m <<= 1) {
    sum += __shfl_xor(sum, m, 32);
    sq  += __shfl_xor(sq,  m, 32);
  }
  const float mu  = sum * (1.f / DII);
  const float var = sq * (1.f / DII) - mu * mu;
  const float is  = rsqrtf(var + 1e-5f);
  const float* zr = zb + (size_t)wave * DII;
  #pragma unroll
  for (int j = 0; j < 4; ++j) {
    const int d = lane + 32 * j;
    const float zv = zr[d];
    const float g = zv / (1.f + __expf(-zv));          // SiLU
    yg[(size_t)wave * DII + d] =
        (_Float16)(((v[j] - mu) * is * lnw[d] + lnb[d]) * g);
  }
}

// ---------------------------------------------------------------------------
// 8) out_proj: out[r, 0:64] = yg[r, 0:128] @ W_out^T.  K=128 -> 4 chunks,
//    N=64 -> 4 n-tiles (16 WMMAs / wave).  W_out staged in LDS by TDM with
//    row stride 129 DWORDs (conflict-free); A operand is already f16.
// ---------------------------------------------------------------------------
__global__ void k_outproj(const _Float16* __restrict__ yg, const float* __restrict__ Wout,
                          float* __restrict__ out) {
  __shared__ float sW[64 * 129];            // 64 rows x (128 + 1 pad) DWORDs
  if (threadIdx.x < 32)
    tdm_load_f32(Wout, (unsigned)(unsigned long long)&sW[0], 64 * 128, /*every 128dw*/6);
  __syncthreads();

  const int tid  = blockIdx.x * blockDim.x + threadIdx.x;
  const int wave = tid >> 5, lane = tid & 31;
  if (wave >= NROWS / 16) return;
  const int mlo = lane & 15, hi = lane >> 4;
  const int row0 = wave * 16, row = row0 + mlo;
  const _Float16* ar = yg + (size_t)row * DII;
  v16h a[4];
  #pragma unroll
  for (int kc = 0; kc < 4; ++kc) {
    const int c0 = kc * 32 + hi * 16;
    #pragma unroll
    for (int e = 0; e < 8; ++e) { a[kc][e] = ar[c0 + e]; a[kc][e + 8] = ar[c0 + 8 + e]; }
  }
  for (int nt = 0; nt < 4; ++nt) {
    const int o = nt * 16 + mlo;
    v8f acc = {};
    #pragma unroll
    for (int kc = 0; kc < 4; ++kc) {
      const float* wr = &sW[o * 129 + kc * 32 + hi * 16];
      v16h bf;
      #pragma unroll
      for (int e = 0; e < 16; ++e) bf[e] = (_Float16)wr[e];
      acc = __builtin_amdgcn_wmma_f32_16x16x32_f16(false, a[kc], false, bf,
                                                   (short)0, acc, false, false);
    }
    #pragma unroll
    for (int r = 0; r < 8; ++r)
      out[(size_t)(row0 + hi * 8 + r) * DMM + o] = acc[r];
  }
}

// ---------------------------------------------------------------------------
extern "C" void kernel_launch(void* const* d_in, const int* in_sizes, int n_in,
                              void* d_out, int out_size, void* d_ws, size_t ws_size,
                              hipStream_t stream) {
  (void)in_sizes; (void)n_in; (void)out_size;
  const float* x    = (const float*)d_in[0];
  const float* Win  = (const float*)d_in[1];
  const float* cw   = (const float*)d_in[2];
  const float* cb   = (const float*)d_in[3];
  const float* xpw  = (const float*)d_in[4];
  const float* dtw  = (const float*)d_in[5];
  const float* dtb  = (const float*)d_in[6];
  const float* Al   = (const float*)d_in[7];
  const float* Dsv  = (const float*)d_in[8];
  const float* lnw  = (const float*)d_in[9];
  const float* lnb  = (const float*)d_in[10];
  const float* Wout = (const float*)d_in[11];
  const int*   freq = (const int*)d_in[12];

  char* ws = (char*)d_ws;
  size_t off = 0;
  auto take = [&](size_t bytes) -> char* {
    char* p = ws + off;
    off += (bytes + 255) & ~(size_t)255;
    return p;
  };
  float*    xc   = (float*)take((size_t)BB * DII * LL * 4);
  float*    zb   = (float*)take((size_t)NROWS * DII * 4);
  float*    xf   = (float*)take((size_t)BB * DII * LL * 4);
  float*    dtsr = (float*)take((size_t)BB * KD * DTRK * LL * 4);
  float*    Bsw  = (float*)take((size_t)BB * KD * LL * DSN * 4);
  float*    Csw  = (float*)take((size_t)BB * KD * LL * DSN * 4);
  float*    dlt  = (float*)take((size_t)BB * KD * DII * LL * 4);
  float*    ydir = (float*)take((size_t)BB * KD * DII * LL * 4);
  int*      invw = (int*)  take((size_t)LL * 4);
  _Float16* yg   = (_Float16*)take((size_t)NROWS * DII * 2);
  if (off > ws_size) return;   // insufficient scratch: nothing safe to do

  const dim3 blk(256);
  k_inproj <<<(NROWS / 16 * 32 + 255) / 256, blk, 0, stream>>>(x, Win, xc, zb);
  k_conv   <<<(BB * DII * LL + 255) / 256, blk, 0, stream>>>(xc, cw, cb, xf);
  k_xdbl   <<<(BB * KD * 3 * (LL / 16) * 32 + 255) / 256, blk, 0, stream>>>(xf, xpw, freq, dtsr, Bsw, Csw);
  k_delta  <<<(BB * KD * DII * LL + 255) / 256, blk, 0, stream>>>(dtsr, dtw, dtb, dlt);
  k_inv    <<<(LL + 255) / 256, blk, 0, stream>>>(freq, invw);
  k_scan   <<<(BB * KD * DII * 32 + 255) / 256, blk, 0, stream>>>(xf, dlt, Bsw, Csw, Al, Dsv, freq, ydir);
  k_lngate <<<(NROWS * 32 + 255) / 256, blk, 0, stream>>>(ydir, zb, invw, lnw, lnb, yg);
  k_outproj<<<(NROWS / 16 * 32 + 255) / 256, blk, 0, stream>>>(yg, Wout, (float*)d_out);
}
